// AffinityLayer_5506148073921
// MI455X (gfx1250) — matrix operand order repaired and checked by reading, compiled, not run on hardware
//
#include <hip/hip_runtime.h>

// ---------------------------------------------------------------------------
// Causal attention (flash-attention style) for gfx1250 (CDNA5, wave32, WMMA)
// B=8, T=2048, D=1024, fp32 in/out; compute via v_wmma_f32_16x16x32_bf16.
//
// Fast path (workspace permitting):
//   pass 1: V [B,T,D] f32  -> Vt [B,D,T] bf16   (transpose + downconvert)
//   pass 2: K [B,T,D] f32  -> Kb [B,T,D] bf16   (downconvert)
//   pass 3: flash attention; every WMMA operand fragment is two contiguous
//           16-byte global_load_b128 of bf16; all K fragments of a key tile
//           preloaded (enforced by a scheduling fence) before the chained
//           S WMMAs; softmax parallelized across all 32 lanes of wave 0.
// ---------------------------------------------------------------------------

typedef __attribute__((ext_vector_type(16))) __bf16 v16bf;
typedef __attribute__((ext_vector_type(8)))  __bf16 v8bf;
typedef __attribute__((ext_vector_type(2)))  __bf16 v2bf;
typedef __attribute__((ext_vector_type(8)))  float  v8f;

#define B_SZ    8
#define T_SEQ   2048
#define D_DIM   1024
#define M_TILE  16      // query rows per workgroup
#define N_TILE  32      // keys per iteration
#define WAVES   8
#define D_SLICE 128     // D / WAVES, per-wave slice of the embedding dim

__device__ __forceinline__ __bf16 f2bf(float f) { return (__bf16)f; }

__device__ __forceinline__ v16bf cat8(v8bf lo, v8bf hi) {
    return __builtin_shufflevector(lo, hi, 0,1,2,3,4,5,6,7,8,9,10,11,12,13,14,15);
}

__device__ __forceinline__ v16bf load_frag16(const __bf16* p) {
    const v8bf* q = (const v8bf*)p;          // two global_load_b128
    return cat8(q[0], q[1]);
}

// Scheduling fence: keep already-issued loads ahead of what follows.
__device__ __forceinline__ void sched_fence() {
    asm volatile("" ::: "memory");
}

// ---- pass 1: V f32 [b][t][d] -> Vt bf16 [b][d][t] --------------------------
__global__ __launch_bounds__(256)
void transpose_v_kernel(const float* __restrict__ V, __bf16* __restrict__ Vt)
{
    __shared__ float tile[64][65];
    const int tid = threadIdx.x;
    const int b  = blockIdx.z;
    const int t0 = blockIdx.y * 64;
    const int d0 = blockIdx.x * 64;

    const int tx = tid & 15;   // d group (float4)
    const int ty = tid >> 4;   // t row 0..15
    #pragma unroll
    for (int r = 0; r < 4; ++r) {
        const int row = ty + r * 16;
        const float4 v = *(const float4*)(V + ((size_t)(b * T_SEQ + t0 + row) * D_DIM
                                               + d0 + tx * 4));
        tile[row][tx * 4 + 0] = v.x;
        tile[row][tx * 4 + 1] = v.y;
        tile[row][tx * 4 + 2] = v.z;
        tile[row][tx * 4 + 3] = v.w;
    }
    __syncthreads();

    #pragma unroll
    for (int i = 0; i < 8; ++i) {
        const int p  = i * 256 + tid;   // lane-consecutive -> coalesced stores
        const int d  = p >> 5;          // 0..63
        const int tp = p & 31;          // t pair index
        v2bf pk;
        pk[0] = f2bf(tile[2 * tp + 0][d]);
        pk[1] = f2bf(tile[2 * tp + 1][d]);
        *(v2bf*)(Vt + ((size_t)(b * D_DIM + d0 + d) * T_SEQ + t0 + 2 * tp)) = pk;
    }
}

// ---- pass 2: K f32 -> bf16 (same layout) -----------------------------------
__global__ __launch_bounds__(256)
void convert_k_kernel(const float* __restrict__ K, __bf16* __restrict__ Kb)
{
    const size_t base = ((size_t)blockIdx.x * 256 + threadIdx.x) * 8;
    const float4 a = *(const float4*)(K + base);
    const float4 c = *(const float4*)(K + base + 4);
    v8bf o;
    o[0] = f2bf(a.x); o[1] = f2bf(a.y); o[2] = f2bf(a.z); o[3] = f2bf(a.w);
    o[4] = f2bf(c.x); o[5] = f2bf(c.y); o[6] = f2bf(c.z); o[7] = f2bf(c.w);
    *(v8bf*)(Kb + base) = o;
}

// ---- pass 3: attention -----------------------------------------------------
template <bool FAST>
__global__ __launch_bounds__(256, 2)
void attn_causal_wmma(const float* __restrict__ Q,
                      const float* __restrict__ K,
                      const float* __restrict__ V,
                      float* __restrict__ Out,
                      const __bf16* __restrict__ Kb,
                      const __bf16* __restrict__ Vt)
{
    __shared__ float  s_S[M_TILE][N_TILE];   // cross-wave score reduction
    __shared__ __bf16 s_P[M_TILE][N_TILE];   // softmax probabilities (bf16)
    __shared__ float  s_corr[M_TILE];        // per-row rescale factor
    __shared__ float  s_invl[M_TILE];        // 1 / row-sum at the end

    const int tid  = threadIdx.x;
    const int wave = tid >> 5;
    const int lane = tid & 31;
    const int half = lane >> 4;              // wave32: two half-lane groups
    const int l16  = lane & 15;

    const int qt = blockIdx.x % (T_SEQ / M_TILE);
    const int b  = blockIdx.x / (T_SEQ / M_TILE);
    const int q0 = qt * M_TILE;
    const int d0 = wave * D_SLICE;

    const float*  Qb  = Q  + (size_t)b * T_SEQ * D_DIM;
    const float*  Kf  = K  + (size_t)b * T_SEQ * D_DIM;
    const float*  Vf  = V  + (size_t)b * T_SEQ * D_DIM;
    const __bf16* Kbb = FAST ? Kb + (size_t)b * T_SEQ * D_DIM : nullptr;
    const __bf16* Vtb = FAST ? Vt + (size_t)b * D_DIM * T_SEQ : nullptr;

    // ---- Q fragments (A layout): lane row m=l16; elems 0..15 -> K=half*16+i.
    // Loaded once, reused for all key tiles.
    v16bf qf[4];
    {
        const float* qrow = Qb + (size_t)(q0 + l16) * D_DIM + d0;
        #pragma unroll
        for (int c = 0; c < 4; ++c) {
            const float* p = qrow + c * 32 + half * 16;
            #pragma unroll
            for (int i = 0; i < 8; ++i) qf[c][i]     = f2bf(p[i]);
            #pragma unroll
            for (int i = 0; i < 8; ++i) qf[c][8 + i] = f2bf(p[8 + i]);
        }
    }

    // O accumulator: 8 N-subtiles of 16x16 over this wave's 128-wide slice
    v8f o[8];
    #pragma unroll
    for (int nj = 0; nj < 8; ++nj)
        #pragma unroll
        for (int r = 0; r < 8; ++r) o[nj][r] = 0.0f;

    float m_i = -__builtin_inff();
    float l_i = 0.0f;

    const int nt_max = (q0 + M_TILE - 1) / N_TILE;

    for (int nt = 0; nt <= nt_max; ++nt) {
        const int k0 = nt * N_TILE;

        if (nt < nt_max) {
            if (FAST)
                __builtin_prefetch(Kbb + (size_t)(k0 + N_TILE + l16) * D_DIM + d0, 0, 1);
            else
                __builtin_prefetch(Kf + (size_t)(k0 + N_TILE + l16) * D_DIM + d0, 0, 1);
        }

        // ---- zero the shared score tile ----
        ((float*)s_S)[tid]       = 0.0f;
        ((float*)s_S)[tid + 256] = 0.0f;
        __syncthreads();

        // ---- partial S = Q_slice @ K_slice^T over this wave's 128-wide D ----
        v8f sacc0, sacc1;
        #pragma unroll
        for (int r = 0; r < 8; ++r) { sacc0[r] = 0.0f; sacc1[r] = 0.0f; }

        if (FAST) {
            // Preload ALL key-tile fragments; the scheduling fence pins the
            // 16 global_load_b128 ahead of the chained WMMAs so they stay in
            // flight under matrix-op execution.
            const __bf16* kr0 = Kbb + (size_t)(k0 + l16)      * D_DIM + d0 + half * 16;
            const __bf16* kr1 = Kbb + (size_t)(k0 + 16 + l16) * D_DIM + d0 + half * 16;
            v16bf kf0[4], kf1[4];
            #pragma unroll
            for (int c = 0; c < 4; ++c) {
                kf0[c] = load_frag16(kr0 + c * 32);
                kf1[c] = load_frag16(kr1 + c * 32);
            }
            sched_fence();
            #pragma unroll
            for (int c = 0; c < 4; ++c) {
                sacc0 = __builtin_amdgcn_wmma_f32_16x16x32_bf16(
                    false, qf[c], false, kf0[c], (short)0, sacc0, false, false);
                sacc1 = __builtin_amdgcn_wmma_f32_16x16x32_bf16(
                    false, qf[c], false, kf1[c], (short)0, sacc1, false, false);
            }
        } else {
            #pragma unroll
            for (int c = 0; c < 4; ++c) {
                #pragma unroll
                for (int ni = 0; ni < 2; ++ni) {
                    const float* kp = Kf + (size_t)(k0 + ni * 16 + l16) * D_DIM
                                         + d0 + c * 32 + half * 16;
                    v16bf bfr;
                    #pragma unroll
                    for (int i = 0; i < 16; ++i) bfr[i] = f2bf(kp[i]);
                    v8f& sa = ni ? sacc1 : sacc0;
                    sa = __builtin_amdgcn_wmma_f32_16x16x32_bf16(
                        false, qf[c], false, bfr, (short)0, sa, false, false);
                }
            }
        }

        // ---- cross-wave reduction (C layout elem r -> m=r+8*half, n=l16) ----
        #pragma unroll
        for (int r = 0; r < 8; ++r) {
            atomicAdd(&s_S[r + half * 8][l16],      sacc0[r]);
            atomicAdd(&s_S[r + half * 8][16 + l16], sacc1[r]);
        }
        __syncthreads();

        // ---- online softmax: wave 0, 2 lanes per row (16 cols each) ----
        if (tid < 32) {
            const int   r     = l16;
            const int   qrow  = q0 + r;
            const int   c0    = half * 16;
            const float scale = 0.03125f;    // 1/sqrt(1024)

            float tmax = -__builtin_inff();
            #pragma unroll
            for (int j = 0; j < 16; ++j) {
                const int col = c0 + j;
                float s = (k0 + col > qrow) ? -__builtin_inff() : s_S[r][col] * scale;
                tmax = fmaxf(tmax, s);
            }
            tmax = fmaxf(tmax, __shfl_xor(tmax, 16, 32));   // combine halves
            const float new_m = fmaxf(m_i, tmax);
            const float corr  = __expf(m_i - new_m);
            float psum = 0.0f;
            #pragma unroll
            for (int j = 0; j < 16; ++j) {
                const int col = c0 + j;
                float s = (k0 + col > qrow) ? -__builtin_inff() : s_S[r][col] * scale;
                float p = __expf(s - new_m);
                psum += p;
                s_P[r][col] = f2bf(p);
            }
            psum += __shfl_xor(psum, 16, 32);               // combine halves
            l_i = l_i * corr + psum;
            m_i = new_m;
            if (half == 0) s_corr[r] = corr;
        }
        __syncthreads();

        // ---- rescale O accumulators ----
        float cr[8];
        #pragma unroll
        for (int r = 0; r < 8; ++r) cr[r] = s_corr[r + half * 8];
        #pragma unroll
        for (int nj = 0; nj < 8; ++nj)
            #pragma unroll
            for (int r = 0; r < 8; ++r) o[nj][r] *= cr[r];

        // ---- P fragment from LDS (A layout) ----
        v16bf pf;
        {
            const __bf16* pp = &s_P[l16][half * 16];
            #pragma unroll
            for (int i = 0; i < 8; ++i) pf[i]     = pp[i];
            #pragma unroll
            for (int i = 0; i < 8; ++i) pf[8 + i] = pp[8 + i];
        }

        // ---- O_slice += P @ V_slice ----
        #pragma unroll
        for (int nj = 0; nj < 8; ++nj) {
            v16bf vfr;
            if (FAST) {
                vfr = load_frag16(Vtb + (size_t)(d0 + nj * 16 + l16) * T_SEQ
                                      + k0 + half * 16);
            } else {
                const float* vp = Vf + (size_t)(k0 + half * 16) * D_DIM
                                     + d0 + nj * 16 + l16;
                #pragma unroll
                for (int i = 0; i < 16; ++i) vfr[i] = f2bf(vp[(size_t)i * D_DIM]);
            }
            o[nj] = __builtin_amdgcn_wmma_f32_16x16x32_bf16(
                false, pf, false, vfr, (short)0, o[nj], false, false);
        }
    }

    // ---- final normalization ----
    if (tid < 32 && half == 0) s_invl[l16] = 1.0f / l_i;
    __syncthreads();

    float il[8];
    #pragma unroll
    for (int r = 0; r < 8; ++r) il[r] = s_invl[r + half * 8];

    float* Ob = Out + (size_t)b * T_SEQ * D_DIM;
    #pragma unroll
    for (int nj = 0; nj < 8; ++nj)
        #pragma unroll
        for (int r = 0; r < 8; ++r)
            Ob[(size_t)(q0 + r + half * 8) * D_DIM + d0 + nj * 16 + l16] =
                o[nj][r] * il[r];
}

extern "C" void kernel_launch(void* const* d_in, const int* in_sizes, int n_in,
                              void* d_out, int out_size, void* d_ws, size_t ws_size,
                              hipStream_t stream) {
    const float* q = (const float*)d_in[0];
    const float* k = (const float*)d_in[1];
    const float* v = (const float*)d_in[2];
    float* out     = (float*)d_out;

    const size_t elems = (size_t)B_SZ * T_SEQ * D_DIM;   // 16,777,216
    const size_t szVt  = elems * sizeof(__bf16);         // 32 MiB
    const size_t szKb  = elems * sizeof(__bf16);         // 32 MiB

    dim3 agrid(B_SZ * (T_SEQ / M_TILE));
    dim3 ablock(256);

    if (ws_size >= szVt + szKb) {
        __bf16* Vt = (__bf16*)d_ws;
        __bf16* Kb = (__bf16*)((char*)d_ws + szVt);

        dim3 tgrid(D_DIM / 64, T_SEQ / 64, B_SZ);
        transpose_v_kernel<<<tgrid, 256, 0, stream>>>(v, Vt);

        dim3 cgrid((unsigned)(elems / (256 * 8)));
        convert_k_kernel<<<cgrid, 256, 0, stream>>>(k, Kb);

        attn_causal_wmma<true><<<agrid, ablock, 0, stream>>>(q, k, v, out, Kb, Vt);
    } else {
        attn_causal_wmma<false><<<agrid, ablock, 0, stream>>>(
            q, k, v, out, nullptr, nullptr);
    }
}
